// TacotronMelDecoder_89859305767376
// MI455X (gfx1250) — matrix operand order repaired and checked by reading, not compiled, not run on hardware
//
#include <hip/hip_runtime.h>
#include <hip/hip_bf16.h>
#include <cstdint>

// ---------------- constants (match reference) ----------------
#define BB 32
#define TT 200
#define SS 256
#define LL 512
#define UU 1024      // 2L
#define EE 512
#define MM 80

typedef __attribute__((ext_vector_type(16))) __bf16 bf16x16;
typedef __attribute__((ext_vector_type(8)))  __bf16 bf16x8;
typedef __attribute__((ext_vector_type(8)))  float  f32x8;

__device__ __forceinline__ unsigned bfb(float f) {
  union { float f; unsigned u; } x; x.f = f;
  return (x.u + 0x7FFFu + ((x.u >> 16) & 1u)) >> 16;    // round-to-nearest-even
}
__device__ __forceinline__ __bf16 f2bf(float f) {
  unsigned short hs = (unsigned short)bfb(f);
  __bf16 out; __builtin_memcpy(&out, &hs, 2);
  return out;
}
__device__ __forceinline__ unsigned pk2(float a, float b) {
  return (bfb(a) & 0xFFFFu) | (bfb(b) << 16);
}
__device__ __forceinline__ float sigm(float x) { return 1.f / (1.f + __expf(-x)); }

// ---------------- A-source / C-sink functors (float4 granularity) ----------------
struct DirectA {
  const float* A; int lda;   // lda multiple of 4
  __device__ float4 load4(int r, int k) const { return *(const float4*)&A[(size_t)r * lda + k]; }
};
// gathers [ x_t(512) | attn_{t-1}(1024) | h_{t-1}(1024) ]; segment bounds are multiples of 4
struct CellinA {
  const float* x; const float* aprev; const float* h; int t;
  __device__ float4 load4(int b, int k) const {
    if (k < LL)      return *(const float4*)&x[((size_t)b * TT + t) * LL + k];
    if (k < LL + UU) return *(const float4*)&aprev[(size_t)b * UU + (k - LL)];
    return *(const float4*)&h[(size_t)b * UU + (k - LL - UU)];
  }
};
struct DirectC {
  float* C; int ldc;
  __device__ float* ptr(int r, int c) const { return C + (size_t)r * ldc + c; }
};
// rows of outs are (t*32+b); mel output is [B,T,M]
struct MelC {
  float* C;
  __device__ float* ptr(int r, int c) const {
    int t = r >> 5, b = r & 31;
    return C + ((size_t)b * TT + t) * MM + c;
  }
};

// ---------------- bf16 WMMA GEMM ----------------
// C[MxN] = act(A[MxK] @ W[KxN] + bias), W pre-transposed+padded: Wt[Npad][Kpad] bf16.
// block = 256 threads (8 waves), tile = 32 rows x 64 cols, K-step 64 (2 WMMA / barrier).
// M must be a multiple of 32. K tail handled by KTAIL (A side) + zero padding (B side).
template <typename AF, typename CF, bool RELU, bool HASBIAS, bool KTAIL, bool NTAIL>
__global__ void __launch_bounds__(256)
gemm_wmma(AF af, CF cf, const __bf16* __restrict__ Wt,
          const float* __restrict__ bias, int M, int N, int K, int Kpad) {
  __shared__ alignas(16) __bf16 As[32 * 72];   // [m][k] bf16, padded row stride 72

  const int tid  = threadIdx.x;
  const int row0 = blockIdx.y * 32;
  const int col0 = blockIdx.x * 64;
  const int wave = tid >> 5, lane = tid & 31;
  const int m0 = (wave >> 2) * 16;             // wave row tile: 0/16
  const int n0 = (wave & 3) * 16;              // wave col tile: 0/16/32/48

  // fragment addressing (CDNA5 16-bit WMMA layouts)
  const int arow = m0 + (lane & 15);
  const int aoff = (lane >> 4) * 8;            // lanes 0-15: K 0-7 & 16-23 ; 16-31: K 8-15 & 24-31
  const int bk   = (lane >> 4) * 8;            // lanes 0-15: K 0-7 & 16-23 ; 16-31: K 8-15 & 24-31
  const __bf16* wp = Wt + (size_t)(col0 + n0 + (lane & 15)) * Kpad + bk;

  // A-fill addressing: thread covers 8 consecutive k of one row
  const int fr = tid >> 3;                     // 0..31
  const int fk = (tid & 7) * 8;                // 0..56

  f32x8 acc = {0.f, 0.f, 0.f, 0.f, 0.f, 0.f, 0.f, 0.f};

  for (int k0 = 0; k0 < K; k0 += 64) {
    // ---- stage A tile (32x64 fp32 -> bf16), 8 elems/thread, one ds_store_b128 ----
    float4 v0, v1;
    const float4 z4 = {0.f, 0.f, 0.f, 0.f};
    if (!KTAIL) {
      v0 = af.load4(row0 + fr, k0 + fk);
      v1 = af.load4(row0 + fr, k0 + fk + 4);
    } else {
      v0 = (k0 + fk + 4 <= K) ? af.load4(row0 + fr, k0 + fk) : z4;
      v1 = (k0 + fk + 8 <= K) ? af.load4(row0 + fr, k0 + fk + 4) : z4;
    }
    uint4 pk;
    pk.x = pk2(v0.x, v0.y); pk.y = pk2(v0.z, v0.w);
    pk.z = pk2(v1.x, v1.y); pk.w = pk2(v1.z, v1.w);
    *(uint4*)&As[fr * 72 + fk] = pk;
    __syncthreads();

#pragma unroll
    for (int sub = 0; sub < 64; sub += 32) {
      const bf16x8 alo = *(const bf16x8*)&As[arow * 72 + sub + aoff];
      const bf16x8 ahi = *(const bf16x8*)&As[arow * 72 + sub + 16 + aoff];
      const bf16x8 blo = *(const bf16x8*)(wp + k0 + sub);
      const bf16x8 bhi = *(const bf16x8*)(wp + k0 + sub + 16);
      const bf16x16 a = __builtin_shufflevector(alo, ahi, 0,1,2,3,4,5,6,7,8,9,10,11,12,13,14,15);
      const bf16x16 b = __builtin_shufflevector(blo, bhi, 0,1,2,3,4,5,6,7,8,9,10,11,12,13,14,15);
      acc = __builtin_amdgcn_wmma_f32_16x16x32_bf16(false, a, false, b, (short)0, acc,
                                                    false, false);
    }
    __syncthreads();
  }

  // ---- store D (f32 16x16 layout: lane<16 rows m0+r, lane>=16 rows m0+8+r) ----
  const int ccol  = col0 + n0 + (lane & 15);
  const int crow0 = row0 + m0 + ((lane >> 4) * 8);
  if (!NTAIL || ccol < N) {
    const float bv = HASBIAS ? bias[ccol] : 0.f;
#pragma unroll
    for (int r = 0; r < 8; ++r) {
      float v = acc[r] + bv;
      if (RELU) v = fmaxf(v, 0.f);
      *cf.ptr(crow0 + r, ccol) = v;
    }
  }
}

// ---------------- pointwise LSTM cell update ----------------
__global__ void lstm_update(const float* __restrict__ z, float* __restrict__ c,
                            float* __restrict__ h, float* __restrict__ ha) {
  const int idx = blockIdx.x * 256 + threadIdx.x;      // 0 .. 32*1024-1
  const int b = idx >> 10, u = idx & 1023;
  const float* zr = z + (size_t)b * (4 * UU);
  const float gi = zr[u], gf = zr[UU + u], gg = zr[2 * UU + u], go = zr[3 * UU + u];
  const float cn = sigm(gf) * c[idx] + sigm(gi) * tanhf(gg);
  const float hn = sigm(go) * tanhf(cn);
  c[idx] = cn;
  h[idx] = hn;
  ha[(size_t)b * (UU + EE) + u] = hn;                  // h half of [h|ctx]
}

// ---------------- Bahdanau attention: score/softmax/context ----------------
__global__ void __launch_bounds__(256)
attention_kernel(const float* __restrict__ keys, const float* __restrict__ q,
                 const float* __restrict__ vatt, const float* __restrict__ memory,
                 float* __restrict__ ha) {
  const int b = blockIdx.x, tid = threadIdx.x;
  __shared__ float qs[UU], vs[UU], red[SS], align[SS];
  for (int u = tid; u < UU; u += 256) { qs[u] = q[(size_t)b * UU + u]; vs[u] = vatt[u]; }
  __syncthreads();

  // score for s = tid
  const float* kr = keys + ((size_t)b * SS + tid) * UU;
  float acc = 0.f;
  for (int u = 0; u < UU; u += 4) {
    const float4 kv = *(const float4*)(kr + u);
    acc += vs[u + 0] * tanhf(kv.x + qs[u + 0]);
    acc += vs[u + 1] * tanhf(kv.y + qs[u + 1]);
    acc += vs[u + 2] * tanhf(kv.z + qs[u + 2]);
    acc += vs[u + 3] * tanhf(kv.w + qs[u + 3]);
  }
  // softmax over 256 encoder steps
  red[tid] = acc; __syncthreads();
  for (int off = 128; off > 0; off >>= 1) {
    if (tid < off) red[tid] = fmaxf(red[tid], red[tid + off]);
    __syncthreads();
  }
  const float mx = red[0]; __syncthreads();
  const float ex = __expf(acc - mx);
  red[tid] = ex; __syncthreads();
  for (int off = 128; off > 0; off >>= 1) {
    if (tid < off) red[tid] += red[tid + off];
    __syncthreads();
  }
  align[tid] = ex / red[0]; __syncthreads();

  // context: ctx[e] = sum_s align[s] * memory[b,s,e]
  for (int e0 = tid; e0 < EE; e0 += 256) {
    float cv = 0.f;
    for (int s = 0; s < SS; ++s)
      cv += align[s] * memory[((size_t)b * SS + s) * EE + e0];
    ha[(size_t)b * (UU + EE) + UU + e0] = cv;          // ctx half of [h|ctx]
  }
}

// ---------------- transpose + convert + pad: Wt[n][k] = bf16(W[k][n]) ----------------
// source rows k < splitK from inA, splitK <= k < K from inB (stacked weights); zero pad.
__global__ void cvtT2(const float* __restrict__ inA, const float* __restrict__ inB,
                      int splitK, __bf16* __restrict__ out,
                      int K, int N, int Kpad, int total) {
  const int i = blockIdx.x * 256 + threadIdx.x;
  if (i >= total) return;
  const int n = i / Kpad, k = i % Kpad;
  float v = 0.f;
  if (n < N) {
    if (k < splitK)      v = inA[(size_t)k * N + n];
    else if (k < K)      v = inB[(size_t)(k - splitK) * N + n];
  }
  out[i] = f2bf(v);
}

// ---------------- utility kernels ----------------
__global__ void zero_f32(float* p, int n) {
  const int i = blockIdx.x * 256 + threadIdx.x;
  if (i < n) p[i] = 0.f;
}
__global__ void copy_tail(const float* __restrict__ h, const float* __restrict__ c,
                          const float* __restrict__ outs, float* __restrict__ dst) {
  const int i = blockIdx.x * 256 + threadIdx.x;        // 0 .. 3*32768-1
  const int NSTATE = BB * UU;
  if (i < NSTATE)            dst[i] = h[i];
  else if (i < 2 * NSTATE)   dst[i] = c[i - NSTATE];
  else if (i < 3 * NSTATE)   dst[i] = outs[(size_t)(TT - 1) * NSTATE + (i - 2 * NSTATE)];
}

// ---------------- host launcher ----------------
extern "C" void kernel_launch(void* const* d_in, const int* in_sizes, int n_in,
                              void* d_out, int out_size, void* d_ws, size_t ws_size,
                              hipStream_t stream) {
  (void)in_sizes; (void)n_in; (void)out_size; (void)ws_size;
  const float* inputs = (const float*)d_in[0];   // [B,T,M]
  const float* memory = (const float*)d_in[1];   // [B,S,E]
  const float* W1  = (const float*)d_in[2];      // [80,1024]
  const float* b1  = (const float*)d_in[3];
  const float* W2  = (const float*)d_in[4];      // [1024,512]
  const float* b2  = (const float*)d_in[5];
  const float* Wx  = (const float*)d_in[6];      // [1536,4096]
  const float* Wh  = (const float*)d_in[7];      // [1024,4096]
  const float* blstm = (const float*)d_in[8];    // [4096]
  const float* Wm  = (const float*)d_in[9];      // [512,1024]
  const float* Wq  = (const float*)d_in[10];     // [1024,1024]
  const float* vat = (const float*)d_in[11];     // [1024]
  const float* Wa  = (const float*)d_in[12];     // [1536,1024]
  const float* Wp  = (const float*)d_in[13];     // [1024,80]
  const float* bp  = (const float*)d_in[14];
  float* out = (float*)d_out;

  // ---- workspace layout ----
  char* ws = (char*)d_ws;
  size_t off = 0;
  auto alloc = [&](size_t bytes) -> void* {
    void* p = ws + off;
    off = (off + bytes + 255) & ~(size_t)255;
    return p;
  };
  const int KZ = LL + UU + UU;                                    // 2560
  // transposed bf16 weights: Wt[Npad][Kpad]
  __bf16* Wzb = (__bf16*)alloc((size_t)4 * UU * KZ * 2);          // [4096][2560]
  __bf16* W1b = (__bf16*)alloc((size_t)2 * LL * 128 * 2);         // [1024][128] (K=80)
  __bf16* W2b = (__bf16*)alloc((size_t)LL * (2 * LL) * 2);        // [512][1024]
  __bf16* Wmb = (__bf16*)alloc((size_t)UU * EE * 2);              // [1024][512]
  __bf16* Wqb = (__bf16*)alloc((size_t)UU * UU * 2);              // [1024][1024]
  __bf16* Wab = (__bf16*)alloc((size_t)UU * (UU + EE) * 2);       // [1024][1536]
  __bf16* Wpb = (__bf16*)alloc((size_t)128 * UU * 2);             // [128][1024] (N=80)
  float* keys = (float*)alloc((size_t)BB * SS * UU * 4);          // [8192,1024]
  float* x    = (float*)alloc((size_t)BB * TT * LL * 4);          // [6400,512]
  float* outs = (float*)alloc((size_t)TT * BB * UU * 4);          // [200,32,1024]; aliases hid1
  float* hid1 = outs;                                             // prenet scratch (same size)
  float* z    = (float*)alloc((size_t)BB * 4 * UU * 4);           // [32,4096]
  float* h    = (float*)alloc((size_t)BB * UU * 4);
  float* c    = (float*)alloc((size_t)BB * UU * 4);
  float* q    = (float*)alloc((size_t)BB * UU * 4);
  float* ha   = (float*)alloc((size_t)BB * (UU + EE) * 4);        // [32,1536] = [h|ctx]
  float* zbuf = (float*)alloc((size_t)BB * UU * 4);               // zeros (attn_{-1})

  auto cvt = [&](const float* srcA, const float* srcB, int splitK, __bf16* dst,
                 int K, int N, int Kpad, int Npad) {
    const int total = Npad * Kpad;
    cvtT2<<<(total + 255) / 256, 256, 0, stream>>>(srcA, srcB, splitK, dst, K, N, Kpad, total);
  };
  // ---- one-time weight transpose/convert to bf16 ----
  cvt(Wx, Wh, LL + UU, Wzb, KZ, 4 * UU, KZ, 4 * UU);   // stacked [Wx;Wh]
  cvt(W1, W1, MM, W1b, MM, 2 * LL, 128, 2 * LL);
  cvt(W2, W2, 2 * LL, W2b, 2 * LL, LL, 2 * LL, LL);
  cvt(Wm, Wm, EE, Wmb, EE, UU, EE, UU);
  cvt(Wq, Wq, UU, Wqb, UU, UU, UU, UU);
  cvt(Wa, Wa, UU + EE, Wab, UU + EE, UU, UU + EE, UU);
  cvt(Wp, Wp, UU, Wpb, UU, MM, UU, 128);

  zero_f32<<<(BB * UU + 255) / 256, 256, 0, stream>>>(h, BB * UU);
  zero_f32<<<(BB * UU + 255) / 256, 256, 0, stream>>>(c, BB * UU);
  zero_f32<<<(BB * UU + 255) / 256, 256, 0, stream>>>(zbuf, BB * UU);

  // ---- prenet: hid1 = relu(inputs @ W1 + b1), x = relu(hid1 @ W2 + b2) ----
  gemm_wmma<DirectA, DirectC, true, true, true, false><<<dim3(16, 200), 256, 0, stream>>>(
      DirectA{inputs, MM}, DirectC{hid1, 2 * LL}, W1b, b1, BB * TT, 2 * LL, MM, 128);
  gemm_wmma<DirectA, DirectC, true, true, false, false><<<dim3(8, 200), 256, 0, stream>>>(
      DirectA{hid1, 2 * LL}, DirectC{x, LL}, W2b, b2, BB * TT, LL, 2 * LL, 2 * LL);

  // ---- keys = memory @ Wm ----
  gemm_wmma<DirectA, DirectC, false, false, false, false><<<dim3(16, 256), 256, 0, stream>>>(
      DirectA{memory, EE}, DirectC{keys, UU}, Wmb, nullptr, BB * SS, UU, EE, EE);

  // ---- sequential decoder ----
  for (int t = 0; t < TT; ++t) {
    const float* aprev = (t == 0) ? zbuf : outs + (size_t)(t - 1) * BB * UU;
    // z = [x_t | attn_{t-1} | h_{t-1}] @ [Wx;Wh] + b_lstm
    gemm_wmma<CellinA, DirectC, false, true, false, false><<<dim3(64, 1), 256, 0, stream>>>(
        CellinA{x, aprev, h, t}, DirectC{z, 4 * UU}, Wzb, blstm, BB, 4 * UU, KZ, KZ);
    lstm_update<<<(BB * UU) / 256, 256, 0, stream>>>(z, c, h, ha);
    // q = h_new @ Wq
    gemm_wmma<DirectA, DirectC, false, false, false, false><<<dim3(16, 1), 256, 0, stream>>>(
        DirectA{h, UU}, DirectC{q, UU}, Wqb, nullptr, BB, UU, UU, UU);
    // score / softmax / context -> ha[:,1024:1536]
    attention_kernel<<<BB, 256, 0, stream>>>(keys, q, vat, memory, ha);
    // attn_t = [h|ctx] @ Wa   (written straight into outs[t])
    gemm_wmma<DirectA, DirectC, false, false, false, false><<<dim3(16, 1), 256, 0, stream>>>(
        DirectA{ha, UU + EE}, DirectC{outs + (size_t)t * BB * UU, UU}, Wab, nullptr,
        BB, UU, UU + EE, UU + EE);
  }

  // ---- mel = outs @ Wp + bp  (rows t*32+b -> out[b,t,:]) ----
  gemm_wmma<DirectA, MelC, false, true, false, true><<<dim3(2, 200), 256, 0, stream>>>(
      DirectA{outs, UU}, MelC{out}, Wpb, bp, BB * TT, MM, UU, UU);

  // ---- (h, c, attn_final) tail of d_out ----
  copy_tail<<<(3 * BB * UU) / 256, 256, 0, stream>>>(h, c, outs, out + (size_t)BB * TT * MM);
}
